// VarAttention_23407571763394
// MI455X (gfx1250) — matrix operand order repaired and checked by reading, not compiled, not run on hardware
//
#include <hip/hip_runtime.h>
#include <hip/hip_bf16.h>

typedef __attribute__((ext_vector_type(16))) _Float16 v16h;
typedef __attribute__((ext_vector_type(8)))  _Float16 v8h;
typedef __attribute__((ext_vector_type(8)))  float    v8f;

#define Bn    16
#define Hh    12
#define Dm    768
#define dh    64
#define Nn    1569          // 1 + F*n
#define Fc    8
#define nTok  196
#define Mrows (Bn * Nn)     // 25104
#define KEYS  197
#define KP    224           // keys padded to 14 WMMA tiles
#define KT_N  14
#define QT_N  13            // ceil(196/16)
#define QSCALE 0.125f       // 64^-0.5

static __device__ __forceinline__ v8f wmma16(v16h a, v16h b, v8f c) {
  return __builtin_amdgcn_wmma_f32_16x16x32_f16(false, a, false, b, (short)0, c,
                                                false, false);
}

// ---------------------------------------------------------------- cvt f32->f16
__global__ void cvt_f32_f16(const float* __restrict__ s, _Float16* __restrict__ d, int n) {
  for (int i = blockIdx.x * blockDim.x + threadIdx.x; i < n; i += gridDim.x * blockDim.x)
    d[i] = (_Float16)s[i];
}

// ---------------------------------------------------------------- QKV GEMM
// C[m, col] = sum_k x[m,k] * Wqkv[col,k]; scatter into head-major q/k/v (f16).
__global__ __launch_bounds__(128) void qkv_gemm(const float* __restrict__ x,
                                                const _Float16* __restrict__ Wh,
                                                _Float16* __restrict__ q_ws,
                                                _Float16* __restrict__ k_ws,
                                                _Float16* __restrict__ v_ws) {
  const int lane = threadIdx.x & 31, wid = threadIdx.x >> 5;
  const int lrow = lane & 15, lhi = lane >> 4;
  const int mBase = blockIdx.x * 64 + wid * 16;
  const int n0 = blockIdx.y * 64;
  int aRow = mBase + lrow;
  if (aRow >= Mrows) aRow = Mrows - 1;
  const int khalf = lhi * 16;

  v8f acc[4] = {};
#pragma unroll 4
  for (int kk = 0; kk < Dm / 32; ++kk) {
    const float* ap = x + (size_t)aRow * Dm + kk * 32 + khalf;
    v8f f0 = *(const v8f*)ap;
    v8f f1 = *(const v8f*)(ap + 8);
    v16h a;
#pragma unroll
    for (int i = 0; i < 8; ++i) { a[i] = (_Float16)f0[i]; a[8 + i] = (_Float16)f1[i]; }
#pragma unroll
    for (int nt = 0; nt < 4; ++nt) {
      const int col = n0 + nt * 16 + lrow;
      const v16h* bp = (const v16h*)(Wh + (size_t)col * Dm + kk * 32 + lhi * 16);
      acc[nt] = wmma16(a, *bp, acc[nt]);
    }
  }
#pragma unroll
  for (int nt = 0; nt < 4; ++nt) {
    const int col = n0 + nt * 16 + lrow;
    const int which = col / Dm, rem = col % Dm;
    const int hd = rem >> 6, dI = rem & 63;
    _Float16* dst = (which == 0) ? q_ws : (which == 1 ? k_ws : v_ws);
    const float sc = (which == 0) ? QSCALE : 1.0f;
#pragma unroll
    for (int i = 0; i < 8; ++i) {
      const int m = mBase + i + lhi * 8;
      if (m < Mrows) {
        const int b = m / Nn, nrow = m % Nn;
        dst[(((size_t)b * Hh + hd) * Nn + nrow) * dh + dI] = (_Float16)(acc[nt][i] * sc);
      }
    }
  }
}

// ---------------------------------------------------------------- block attention
__global__ __launch_bounds__(128) void attn_block(const _Float16* __restrict__ q_ws,
                                                  const _Float16* __restrict__ k_ws,
                                                  const _Float16* __restrict__ v_ws,
                                                  const float* __restrict__ x_mask,
                                                  _Float16* __restrict__ attn_ws) {
  __shared__ alignas(32) _Float16 VT[dh][KP];        // V transposed: [d][key]
  __shared__ alignas(32) _Float16 Pbuf[4][16][KP];   // per-wave P staging
  __shared__ float maskS[KP];

  const int bid = blockIdx.x;
  const int bh = bid / Fc, f = bid % Fc;
  const int b = bh / Hh, hd = bh % Hh;
  const size_t headBase = (size_t)bh * Nn * dh;
  const int tid = threadIdx.x;

  for (int idx = tid; idx < dh * KP; idx += 128) {
    const int dI = idx / KP, kI = idx % KP;
    _Float16 val = (_Float16)0.0f;
    if (kI < KEYS) {
      const int krow = (kI == 0) ? 0 : (1 + f * nTok + kI - 1);
      val = v_ws[headBase + (size_t)krow * dh + dI];
    }
    VT[dI][kI] = val;
  }
  if (tid < KP) {
    float mv = -1e30f;
    if (tid < KEYS) {
      const int krow = (tid == 0) ? 0 : (1 + f * nTok + tid - 1);
      mv = x_mask[(size_t)b * Nn + krow];
    }
    maskS[tid] = mv;
  }
  __syncthreads();

  const int lane = tid & 31, wid = tid >> 5;
  const int lrow = lane & 15, lhi = lane >> 4;

  for (int qt = wid; qt < QT_N; qt += 4) {
    const int q0 = qt * 16;
    int qrow = q0 + lrow;
    if (qrow > nTok - 1) qrow = nTok - 1;
    const size_t qoff = headBase + (size_t)(1 + f * nTok + qrow) * dh;

    v8f S[KT_N] = {};
#pragma unroll
    for (int kk = 0; kk < 2; ++kk) {
      const _Float16* ap = q_ws + qoff + kk * 32 + lhi * 16;
      v8h a0 = *(const v8h*)ap;
      v8h a1 = *(const v8h*)(ap + 8);
      v16h a;
#pragma unroll
      for (int i = 0; i < 8; ++i) { a[i] = a0[i]; a[8 + i] = a1[i]; }
#pragma unroll
      for (int kt = 0; kt < KT_N; ++kt) {
        int j = kt * 16 + lrow;
        if (j > KEYS - 1) j = KEYS - 1;                 // clamped loads, masked later
        const int krow = (j == 0) ? 0 : (1 + f * nTok + j - 1);
        const v16h* bp =
            (const v16h*)(k_ws + headBase + (size_t)krow * dh + kk * 32 + lhi * 16);
        S[kt] = wmma16(a, *bp, S[kt]);
      }
    }

    // additive mask (also kills padded key columns)
#pragma unroll
    for (int kt = 0; kt < KT_N; ++kt) {
      const float mval = maskS[kt * 16 + lrow];
#pragma unroll
      for (int i = 0; i < 8; ++i) S[kt][i] += mval;
    }

    // row softmax; rows live in element index + lane-half, cols across 16 lanes
    float mx[8], sm[8];
#pragma unroll
    for (int i = 0; i < 8; ++i) mx[i] = -3.0e38f;
#pragma unroll
    for (int kt = 0; kt < KT_N; ++kt)
#pragma unroll
      for (int i = 0; i < 8; ++i) mx[i] = fmaxf(mx[i], S[kt][i]);
#pragma unroll
    for (int sh = 1; sh < 16; sh <<= 1)
#pragma unroll
      for (int i = 0; i < 8; ++i) mx[i] = fmaxf(mx[i], __shfl_xor(mx[i], sh, 32));
#pragma unroll
    for (int i = 0; i < 8; ++i) sm[i] = 0.0f;
#pragma unroll
    for (int kt = 0; kt < KT_N; ++kt)
#pragma unroll
      for (int i = 0; i < 8; ++i) {
        const float e = __expf(S[kt][i] - mx[i]);
        S[kt][i] = e;
        sm[i] += e;
      }
#pragma unroll
    for (int sh = 1; sh < 16; sh <<= 1)
#pragma unroll
      for (int i = 0; i < 8; ++i) sm[i] += __shfl_xor(sm[i], sh, 32);
    float inv[8];
#pragma unroll
    for (int i = 0; i < 8; ++i) inv[i] = 1.0f / sm[i];

    // stage P (C-layout -> LDS row-major), then read back as A fragments
#pragma unroll
    for (int kt = 0; kt < KT_N; ++kt)
#pragma unroll
      for (int i = 0; i < 8; ++i)
        Pbuf[wid][i + lhi * 8][kt * 16 + lrow] = (_Float16)(S[kt][i] * inv[i]);

    v8f O[4] = {};
#pragma unroll
    for (int kk2 = 0; kk2 < KP / 32; ++kk2) {
      const _Float16* pp = &Pbuf[wid][lrow][kk2 * 32 + lhi * 16];
      v8h p0 = *(const v8h*)pp;
      v8h p1 = *(const v8h*)(pp + 8);
      v16h aP;
#pragma unroll
      for (int i = 0; i < 8; ++i) { aP[i] = p0[i]; aP[8 + i] = p1[i]; }
#pragma unroll
      for (int nt = 0; nt < 4; ++nt) {
        const v16h* vp = (const v16h*)&VT[nt * 16 + lrow][kk2 * 32 + lhi * 16];
        O[nt] = wmma16(aP, *vp, O[nt]);
      }
    }

#pragma unroll
    for (int nt = 0; nt < 4; ++nt)
#pragma unroll
      for (int i = 0; i < 8; ++i) {
        const int r = q0 + i + lhi * 8;
        if (r < nTok) {
          const int nrow = 1 + f * nTok + r;
          attn_ws[((size_t)b * Nn + nrow) * Dm + hd * dh + nt * 16 + lrow] =
              (_Float16)O[nt][i];
        }
      }
  }
}

// ---------------------------------------------------------------- cls-token attention
__global__ __launch_bounds__(256) void cls_attn(const _Float16* __restrict__ q_ws,
                                                const _Float16* __restrict__ k_ws,
                                                const _Float16* __restrict__ v_ws,
                                                const float* __restrict__ x_mask,
                                                _Float16* __restrict__ attn_ws) {
  __shared__ float sbuf[Nn];
  __shared__ float red[256];
  __shared__ float psum[4][dh];
  __shared__ float qsh[dh];

  const int bh = blockIdx.x, b = bh / Hh, hd = bh % Hh;
  const size_t base = (size_t)bh * Nn * dh;
  const int tid = threadIdx.x;

  if (tid < dh) qsh[tid] = (float)q_ws[base + tid];
  __syncthreads();

  float lmax = -3.0e38f;
  for (int j = tid; j < Nn; j += 256) {
    const _Float16* kr = k_ws + base + (size_t)j * dh;
    float s = 0.0f;
#pragma unroll 16
    for (int d = 0; d < dh; ++d) s += qsh[d] * (float)kr[d];
    s += x_mask[(size_t)b * Nn + j];
    sbuf[j] = s;
    lmax = fmaxf(lmax, s);
  }
  red[tid] = lmax;
  __syncthreads();
  for (int off = 128; off > 0; off >>= 1) {
    if (tid < off) red[tid] = fmaxf(red[tid], red[tid + off]);
    __syncthreads();
  }
  const float m = red[0];
  float lsum = 0.0f;
  for (int j = tid; j < Nn; j += 256) {
    const float e = __expf(sbuf[j] - m);
    sbuf[j] = e;
    lsum += e;
  }
  __syncthreads();
  red[tid] = lsum;
  __syncthreads();
  for (int off = 128; off > 0; off >>= 1) {
    if (tid < off) red[tid] += red[tid + off];
    __syncthreads();
  }
  const float inv = 1.0f / red[0];

  const int c = tid >> 6, dI = tid & 63;
  float acc = 0.0f;
  for (int j = c; j < Nn; j += 4) acc += sbuf[j] * (float)v_ws[base + (size_t)j * dh + dI];
  psum[c][dI] = acc;
  __syncthreads();
  if (tid < dh) {
    const float o = (psum[0][tid] + psum[1][tid] + psum[2][tid] + psum[3][tid]) * inv;
    attn_ws[((size_t)b * Nn) * Dm + hd * dh + tid] = (_Float16)o;
  }
}

// ---------------------------------------------------------------- output projection
__global__ __launch_bounds__(128) void proj_gemm(const _Float16* __restrict__ A,
                                                 const _Float16* __restrict__ Wh,
                                                 const float* __restrict__ bias,
                                                 float* __restrict__ out) {
  const int lane = threadIdx.x & 31, wid = threadIdx.x >> 5;
  const int lrow = lane & 15, lhi = lane >> 4;
  const int mBase = blockIdx.x * 64 + wid * 16;
  const int n0 = blockIdx.y * 64;
  int aRow = mBase + lrow;
  if (aRow >= Mrows) aRow = Mrows - 1;

  v8f acc[4] = {};
#pragma unroll 4
  for (int kk = 0; kk < Dm / 32; ++kk) {
    const _Float16* ap = A + (size_t)aRow * Dm + kk * 32 + lhi * 16;
    v8h a0 = *(const v8h*)ap;
    v8h a1 = *(const v8h*)(ap + 8);
    v16h a;
#pragma unroll
    for (int i = 0; i < 8; ++i) { a[i] = a0[i]; a[8 + i] = a1[i]; }
#pragma unroll
    for (int nt = 0; nt < 4; ++nt) {
      const int col = n0 + nt * 16 + lrow;
      const v16h* bp = (const v16h*)(Wh + (size_t)col * Dm + kk * 32 + lhi * 16);
      acc[nt] = wmma16(a, *bp, acc[nt]);
    }
  }
#pragma unroll
  for (int nt = 0; nt < 4; ++nt) {
    const int col = n0 + nt * 16 + lrow;
    const float bv = bias[col];
#pragma unroll
    for (int i = 0; i < 8; ++i) {
      const int m = mBase + i + lhi * 8;
      if (m < Mrows) out[(size_t)m * Dm + col] = acc[nt][i] + bv;
    }
  }
}

// ---------------------------------------------------------------- launch
extern "C" void kernel_launch(void* const* d_in, const int* in_sizes, int n_in,
                              void* d_out, int out_size, void* d_ws, size_t ws_size,
                              hipStream_t stream) {
  const float* x      = (const float*)d_in[0];   // (B, N, D)
  const float* x_mask = (const float*)d_in[1];   // (B, N)
  const float* Wqkv   = (const float*)d_in[2];   // (3D, D)
  const float* Wproj  = (const float*)d_in[3];   // (D, D)
  const float* bproj  = (const float*)d_in[4];   // (D,)
  float* out = (float*)d_out;
  (void)in_sizes; (void)n_in; (void)out_size; (void)ws_size;

  // workspace carve (all 256B aligned)
  char* ws = (char*)d_ws;
  const size_t szWqkv = (size_t)3 * Dm * Dm * sizeof(_Float16);   //  3.5 MB
  const size_t szWprj = (size_t)Dm * Dm * sizeof(_Float16);       //  1.2 MB
  const size_t szHead = (size_t)Bn * Hh * Nn * dh * sizeof(_Float16); // 38.6 MB
  const size_t szAttn = (size_t)Mrows * Dm * sizeof(_Float16);    // 38.6 MB
  _Float16* wqkvh = (_Float16*)(ws);
  _Float16* wprjh = (_Float16*)(ws + szWqkv);
  _Float16* q_ws  = (_Float16*)(ws + szWqkv + szWprj);
  _Float16* k_ws  = (_Float16*)(ws + szWqkv + szWprj + szHead);
  _Float16* v_ws  = (_Float16*)(ws + szWqkv + szWprj + 2 * szHead);
  _Float16* a_ws  = (_Float16*)(ws + szWqkv + szWprj + 3 * szHead);
  (void)szAttn;

  cvt_f32_f16<<<512, 256, 0, stream>>>(Wqkv, wqkvh, 3 * Dm * Dm);
  cvt_f32_f16<<<512, 256, 0, stream>>>(Wproj, wprjh, Dm * Dm);

  dim3 gq((Mrows + 63) / 64, (3 * Dm) / 64);    // (393, 36)
  qkv_gemm<<<gq, 128, 0, stream>>>(x, wqkvh, q_ws, k_ws, v_ws);

  cls_attn<<<Bn * Hh, 256, 0, stream>>>(q_ws, k_ws, v_ws, x_mask, a_ws);
  attn_block<<<Bn * Hh * Fc, 128, 0, stream>>>(q_ws, k_ws, v_ws, x_mask, a_ws);

  dim3 gp((Mrows + 63) / 64, Dm / 64);          // (393, 12)
  proj_gemm<<<gp, 128, 0, stream>>>(a_ws, wprjh, bproj, out);
}
